// TTCNPatchEncoder_79946521247943
// MI455X (gfx1250) — matrix operand order repaired and checked by reading, not compiled, not run on hardware
//
#include <hip/hip_runtime.h>

#define LMAX   128
#define IN_DIM 32
#define TTCN   32
#define OUT_DIM 33
#define FDIM   (IN_DIM * TTCN)  // 1024
#define NEGV   (-1.0e8f)
#define LN_EPS 1e-5f

typedef __attribute__((ext_vector_type(16))) __bf16 v16bf;
typedef __attribute__((ext_vector_type(8)))  __bf16 v8bf;
typedef __attribute__((ext_vector_type(8)))  float  v8f;

struct __align__(16) Smem {
  __bf16 xb[LMAX * IN_DIM];   // x, bf16, row-major (A operands)
  float  xf[LMAX * IN_DIM];   // x, f32, for the einsum contraction
  __bf16 w1t[TTCN * IN_DIM];  // W1 column-major [j][k] (B operands)
  __bf16 w2t[TTCN * TTCN];    // W2 column-major
  __bf16 w3t[FDIM * TTCN];    // W3 column-major [j][k]
  __bf16 h1b[LMAX * TTCN];    // relu(x@W1+b1), bf16 row-major
  __bf16 h2b[LMAX * TTCN];    // relu(h1@W2+b2), bf16 row-major
  float  maskb[LMAX];
  float  b1s[TTCN];
  float  b2s[TTCN];
  float  b3s[FDIM];
  float  biass[TTCN];
  float  sbuf[FDIM];          // per-column softmax-weighted dot with x
};

// native bf16 conversion (lowers to the HW cvt path, RTNE)
__device__ __forceinline__ unsigned pack2_bf16(float a, float b) {
  union { __bf16 h[2]; unsigned u; } o;
  o.h[0] = (__bf16)a;
  o.h[1] = (__bf16)b;
  return o.u;
}

__device__ __forceinline__ v8f wmma_bf16(v16bf a, v16bf b, v8f c) {
  // D = A(16x32 bf16) * B(32x16 bf16) + C(16x16 f32)
  return __builtin_amdgcn_wmma_f32_16x16x32_bf16(false, a, false, b, (short)0, c,
                                                 false, false);
}

// A fragment: 16x32 bf16 from row-major LDS (ld=32).
// ISA layout: lane<16 -> K {0..7, 16..23}; lane>=16 -> K {8..15, 24..31}; M = row0 + lane%16.
__device__ __forceinline__ v16bf load_a_frag(const __bf16* base, int row0, int lane) {
  int m  = row0 + (lane & 15);
  int k0 = (lane < 16) ? 0 : 8;
  const __bf16* p = base + m * 32 + k0;
  v8bf lo = *(const v8bf*)(p);        // K = k0 .. k0+7   (16B aligned)
  v8bf hi = *(const v8bf*)(p + 16);   // K = k0+16 .. k0+23
  v16bf a;
#pragma unroll
  for (int i = 0; i < 8; ++i) { a[i] = lo[i]; a[i + 8] = hi[i]; }
  return a;
}

// B fragment: 32x16 bf16 from column-major LDS (K contiguous, ld=32).
// ISA layout: lane<16 -> K {0..7, 16..23}; lane>=16 -> K {8..15, 24..31}; N = col0 + lane%16.
__device__ __forceinline__ v16bf load_b_frag(const __bf16* colmajor, int col0, int lane) {
  int nn = col0 + (lane & 15);
  int k0 = (lane < 16) ? 0 : 8;
  const __bf16* p = colmajor + nn * 32 + k0;
  v8bf lo = *(const v8bf*)(p);        // K = k0 .. k0+7
  v8bf hi = *(const v8bf*)(p + 16);   // K = k0+16 .. k0+23
  v16bf b;
#pragma unroll
  for (int i = 0; i < 8; ++i) { b[i] = lo[i]; b[i + 8] = hi[i]; }
  return b;
}

// C/D layout: VGPR r holds M = row0 + r (lanes 0-15) / row0 + 8 + r (lanes 16-31), N = lane%16.
__device__ __forceinline__ void store_cd_relu_bf16(__bf16* dst, int row0, int col0,
                                                   int lane, v8f c, const float* bvec) {
  int col = col0 + (lane & 15);
  float bb = bvec[col];
  int r0 = row0 + ((lane >= 16) ? 8 : 0);
#pragma unroll
  for (int r = 0; r < 8; ++r) {
    float v = fmaxf(c[r] + bb, 0.0f);
    dst[(r0 + r) * 32 + col] = (__bf16)v;
  }
}

__global__ void __launch_bounds__(256)
ttcn_patch_encoder_kernel(const float* __restrict__ x, const float* __restrict__ mask,
                          const float* __restrict__ W1, const float* __restrict__ b1,
                          const float* __restrict__ W2, const float* __restrict__ b2,
                          const float* __restrict__ W3, const float* __restrict__ b3,
                          const float* __restrict__ bias, const float* __restrict__ gamma,
                          const float* __restrict__ beta, float* __restrict__ out) {
  extern __shared__ char smem_raw[];
  Smem& S = *reinterpret_cast<Smem*>(smem_raw);

  const int tid  = threadIdx.x;
  const int wave = tid >> 5;
  const int lane = tid & 31;
  const int n    = blockIdx.x;

  // ---------------- stage 0: cooperative loads into LDS (pair-packed bf16) --------
  const float2* xg2 = (const float2*)(x + (size_t)n * (LMAX * IN_DIM));
#pragma unroll
  for (int it = 0; it < (LMAX * IN_DIM) / 512; ++it) {   // 2048 float2 pairs
    int p = tid + it * 256;
    float2 v = xg2[p];
    ((float2*)S.xf)[p] = v;
    *(unsigned*)&S.xb[2 * p] = pack2_bf16(v.x, v.y);
  }
  if (tid < LMAX) S.maskb[tid] = mask[(size_t)n * LMAX + tid];
  // W1/W2 transpose: thread owns rows (k, k+1) of column j -> one packed store
#pragma unroll
  for (int it = 0; it < (IN_DIM * TTCN) / 512; ++it) {   // 512 pairs
    int p = tid + it * 256;
    int j = p & 31, k = (p >> 5) * 2;
    S.w1t[j * IN_DIM + k] = (__bf16)0.0f;  // placeholder overwritten below (keeps flow simple)
    *(unsigned*)&S.w1t[j * IN_DIM + k] = pack2_bf16(W1[k * TTCN + j], W1[(k + 1) * TTCN + j]);
    *(unsigned*)&S.w2t[j * TTCN  + k] = pack2_bf16(W2[k * TTCN + j], W2[(k + 1) * TTCN + j]);
  }
  // W3 transpose: 16384 pairs; j coalesced across lanes, packed b32 LDS stores
#pragma unroll
  for (int it = 0; it < (TTCN * FDIM) / 512; ++it) {
    int p = tid + it * 256;
    int j = p & (FDIM - 1), k = (p >> 10) * 2;
    *(unsigned*)&S.w3t[j * TTCN + k] =
        pack2_bf16(W3[k * FDIM + j], W3[(k + 1) * FDIM + j]);
  }
  for (int idx = tid; idx < FDIM; idx += 256) S.b3s[idx] = b3[idx];
  if (tid < TTCN) { S.b1s[tid] = b1[tid]; S.b2s[tid] = b2[tid]; S.biass[tid] = bias[tid]; }
  __syncthreads();

  // ---------------- stage 1: h1 = relu(x @ W1 + b1), wave owns rows 16w..16w+15 ----
  {
    v16bf a   = load_a_frag(S.xb, wave * 16, lane);
    v16bf b0  = load_b_frag(S.w1t, 0, lane);
    v16bf b1f = load_b_frag(S.w1t, 16, lane);
    v8f z0 = {0,0,0,0,0,0,0,0}, z1 = {0,0,0,0,0,0,0,0};
    v8f c0 = wmma_bf16(a, b0, z0);
    v8f c1 = wmma_bf16(a, b1f, z1);
    store_cd_relu_bf16(S.h1b, wave * 16, 0, lane, c0, S.b1s);
    store_cd_relu_bf16(S.h1b, wave * 16, 16, lane, c1, S.b1s);
  }
  __syncthreads();

  // ---------------- stage 2: h2 = relu(h1 @ W2 + b2) ----------------
  {
    v16bf a   = load_a_frag(S.h1b, wave * 16, lane);
    v16bf b0  = load_b_frag(S.w2t, 0, lane);
    v16bf b1f = load_b_frag(S.w2t, 16, lane);
    v8f z0 = {0,0,0,0,0,0,0,0}, z1 = {0,0,0,0,0,0,0,0};
    v8f c0 = wmma_bf16(a, b0, z0);
    v8f c1 = wmma_bf16(a, b1f, z1);
    store_cd_relu_bf16(S.h2b, wave * 16, 0, lane, c0, S.b2s);
    store_cd_relu_bf16(S.h2b, wave * 16, 16, lane, c1, S.b2s);
  }
  __syncthreads();

  // ---------------- stage 3: f = h2 @ W3 + b3; masked softmax over L; dot with x ----
  v16bf A[8];
#pragma unroll
  for (int t = 0; t < 8; ++t) A[t] = load_a_frag(S.h2b, t * 16, lane);
  const int lane16 = lane & 15;
  const int rowadd = (lane < 16) ? 0 : 8;

  for (int nt = 0; nt < 8; ++nt) {
    const int col0 = (wave * 8 + nt) * 16;   // wave owns columns [128w, 128w+128)
    const int j = col0 + lane16;             // this lane-pair's column of f
    v16bf B = load_b_frag(S.w3t, col0, lane);
    v8f acc[8];
#pragma unroll
    for (int t = 0; t < 8; ++t) {
      v8f z = {0,0,0,0,0,0,0,0};
      acc[t] = wmma_bf16(A[t], B, z);        // logits for rows 16t..16t+15
    }
    const float b3j = S.b3s[j];
    // mask + bias: v = (logit+b3j)*mk + (1-mk)*NEG, with (1-mk)*NEG = fma(mk, 1e8, NEG)
    float mx = -3.0e38f;
#pragma unroll
    for (int t = 0; t < 8; ++t)
#pragma unroll
      for (int r = 0; r < 8; ++r) {
        float mk = S.maskb[t * 16 + r + rowadd];
        float v = fmaf(acc[t][r] + b3j, mk, fmaf(mk, 1.0e8f, NEGV));
        acc[t][r] = v;
        mx = fmaxf(mx, v);
      }
    mx = fmaxf(mx, __shfl_xor(mx, 16));
    float ssum = 0.0f;
#pragma unroll
    for (int t = 0; t < 8; ++t)
#pragma unroll
      for (int r = 0; r < 8; ++r) {
        float e = __expf(acc[t][r] - mx);
        acc[t][r] = e;
        ssum += e;
      }
    ssum += __shfl_xor(ssum, 16);
    const float inv = 1.0f / ssum;
    // einsum: s[j] = sum_l softmax(f)[l, j] * x[l, j % 32]
    const int i = j & (IN_DIM - 1);
    float part = 0.0f;
#pragma unroll
    for (int t = 0; t < 8; ++t)
#pragma unroll
      for (int r = 0; r < 8; ++r) {
        int l = t * 16 + r + rowadd;
        part = fmaf(acc[t][r], S.xf[l * IN_DIM + i], part);
      }
    part *= inv;
    part += __shfl_xor(part, 16);
    if (lane < 16) S.sbuf[j] = part;
  }
  __syncthreads();

  // ---------------- stage 4: channel reduce + bias + relu + exists + LayerNorm ----
  if (wave == 0) {
    float t = 0.0f;
#pragma unroll
    for (int k = 0; k < TTCN; ++k) t += S.sbuf[lane * TTCN + k];   // ttcn_out[c]
    float feat = fmaxf(t + S.biass[lane], 0.0f);
    feat = fminf(feat, 1.0e4f);
    if (!(feat == feat)) feat = 0.0f;           // nan_to_num
    float msum = 0.0f;
#pragma unroll
    for (int k = 0; k < LMAX / 32; ++k) msum += S.maskb[lane + 32 * k];
#pragma unroll
    for (int off = 16; off > 0; off >>= 1) msum += __shfl_xor(msum, off);
    const float exists = (msum > 0.0f) ? 1.0f : 0.0f;
    float ssum = feat;
#pragma unroll
    for (int off = 16; off > 0; off >>= 1) ssum += __shfl_xor(ssum, off);
    ssum += exists;
    const float mu = ssum / 33.0f;
    const float d = feat - mu;
    float vs = d * d;
#pragma unroll
    for (int off = 16; off > 0; off >>= 1) vs += __shfl_xor(vs, off);
    const float de = exists - mu;
    vs += de * de;
    const float rs = rsqrtf(vs / 33.0f + LN_EPS);
    float* op = out + (size_t)n * OUT_DIM;
    op[lane] = gamma[lane] * d * rs + beta[lane];
    if (lane == 0) op[32] = gamma[32] * de * rs + beta[32];
  }
}

extern "C" void kernel_launch(void* const* d_in, const int* in_sizes, int n_in,
                              void* d_out, int out_size, void* d_ws, size_t ws_size,
                              hipStream_t stream) {
  (void)n_in; (void)out_size; (void)d_ws; (void)ws_size;
  const float* x    = (const float*)d_in[0];
  const float* mask = (const float*)d_in[1];
  const float* W1   = (const float*)d_in[2];
  const float* b1   = (const float*)d_in[3];
  const float* W2   = (const float*)d_in[4];
  const float* b2   = (const float*)d_in[5];
  const float* W3   = (const float*)d_in[6];
  const float* b3   = (const float*)d_in[7];
  const float* bias = (const float*)d_in[8];
  const float* gamma= (const float*)d_in[9];
  const float* beta = (const float*)d_in[10];
  float* out = (float*)d_out;

  const int n = in_sizes[0] / (LMAX * IN_DIM);   // 1024 samples
  ttcn_patch_encoder_kernel<<<dim3(n), dim3(256), sizeof(Smem), stream>>>(
      x, mask, W1, b1, W2, b2, W3, b3, bias, gamma, beta, out);
}
